// BilinearUpsampler_54829552500772
// MI455X (gfx1250) — matrix operand order repaired and checked by reading, not compiled, not run on hardware
//
#include <hip/hip_runtime.h>
#include <hip/hip_bf16.h>

// ---------------------------------------------------------------------------
// Fused kernel-prediction-network upsampler for MI455X (gfx1250, wave32).
//
// Per-pixel MLP (3->64->256->576) on poseMap via v_wmma_f32_16x16x32_f16,
// softmax over 9 taps, gather-weighted 3x3 accumulation from L2-resident x.
// Fully fused: the 604MB `sw` tensor never touches HBM (~140MB total traffic).
//
// Round-2 change: 32 pixels (2 N-tiles) per workgroup. Each W3 A-fragment
// loaded from L2 is reused across 2 B-tiles -> halves L2 weight traffic
// (4.7GB -> 2.4GB) and gives two independent WMMA accumulator chains per
// M-tile (breaks the RAW-serialized k-loop). LDS grows to ~96KB, so it is
// allocated dynamically (CDNA5 allows up to 320KB per workgroup).
// ---------------------------------------------------------------------------

typedef __attribute__((ext_vector_type(16))) _Float16 v16h;
typedef __attribute__((ext_vector_type(8)))  float    v8f;

#define HO    512
#define WO    512
#define NPIX  (HO * WO)      // 262144 pixels
#define HI    128
#define WI    128
#define CCH   64             // channels
#define M2    256            // GEMM2 rows  (W2: 256x64)
#define M3    576            // GEMM3 rows  (W3: 576x256)

#define NPX   32             // pixels per workgroup (2 WMMA N-tiles)

// LDS row strides (elements): multiples of 8 halfs (16B alignment for b128
// fragment loads), padded to break bank-conflict phase.
#define LD1   80             // h1  rows: 64 + 16 halfs
#define LD2   272            // h2  rows: 256 + 16 halfs
#define LDW   577            // w   rows: 576 + 1 floats

// Dynamic LDS layout (bytes)
#define OFF_H1   0                                   // 32*80*2  = 5120
#define OFF_H2   (OFF_H1 + NPX * LD1 * 2)            // 5120  (16B aligned)
#define OFF_W    (OFF_H2 + NPX * LD2 * 2)            // 22528 (16B aligned)
#define SMEM_SZ  (OFF_W + NPX * LDW * 4)             // 96384 bytes

union AFragU { v16h v; _Float16 h[16]; uint4 q[2]; };
union CFragU { v8f  v; float f[8]; };

// A-matrix fragment (16x32 f16, MxK) from row-major f16 weights.
// ISA layout: lane&15 = M; element i holds K = 16*(i>>3) + 8*(lane>>4) + (i&7)
__device__ __forceinline__ v16h load_a_frag(const _Float16* __restrict__ W,
                                            int ldk, int m0, int k0, int lane) {
    const _Float16* base = W + (m0 + (lane & 15)) * ldk + k0 + ((lane >> 4) << 3);
    AFragU u;
    u.q[0] = *(const uint4*)(base);
    u.q[1] = *(const uint4*)(base + 16);
    return u.v;
}

// B-matrix fragment (32x16 f16, KxN) from LDS staged as [pixel][K] f16.
// ISA layout: lane&15 = N; element i holds K = 16*(i>>3) + 8*(lane>>4) + (i&7)
__device__ __forceinline__ v16h load_b_frag(const _Float16* __restrict__ hB,
                                            int ldk, int n0, int k0, int lane) {
    const _Float16* base = hB + (n0 + (lane & 15)) * ldk + k0 + ((lane >> 4) << 3);
    AFragU u;
    u.q[0] = *(const uint4*)(base);
    u.q[1] = *(const uint4*)(base + 16);
    return u.v;
}

// --------------------------- pre-pass: f32 -> f16 weights -------------------
__global__ __launch_bounds__(256) void kpn_convert_weights(
    const float* __restrict__ W2, const float* __restrict__ W3,
    _Float16* __restrict__ W2h, _Float16* __restrict__ W3h) {
    int i = blockIdx.x * 256 + threadIdx.x;
    if (i < M2 * CCH) W2h[i] = (_Float16)W2[i];
    if (i < M3 * M2)  W3h[i] = (_Float16)W3[i];
}

// --------------------------- fused main kernel ------------------------------
// 256 threads (8 waves) per workgroup; each workgroup owns 32 pixels.
__global__ __launch_bounds__(256) void kpn_fused_kernel(
    const float* __restrict__ x,        // (2,64,128,128)
    const float* __restrict__ poseMap,  // (1,3,512,512)
    const float* __restrict__ W1,       // (64,3)
    const float* __restrict__ b1,       // (64,)
    const float* __restrict__ b2,       // (256,)
    const float* __restrict__ b3,       // (576,)
    const _Float16* __restrict__ W2h,   // (256,64) f16
    const _Float16* __restrict__ W3h,   // (576,256) f16
    const int* __restrict__ mapY,       // (512,512)
    const int* __restrict__ mapX,       // (512,512)
    float* __restrict__ out)            // (2,64,512,512)
{
    extern __shared__ __align__(16) char kpn_smem[];
    _Float16* hB1  = (_Float16*)(kpn_smem + OFF_H1); // [NPX][LD1]
    _Float16* hB2  = (_Float16*)(kpn_smem + OFF_H2); // [NPX][LD2]
    float*    wbuf = (float*)   (kpn_smem + OFF_W);  // [NPX][LDW]

    const int tid  = threadIdx.x;
    const int lane = tid & 31;
    const int wv   = tid >> 5;          // wave id 0..7
    const int pix0 = blockIdx.x * NPX;  // first pixel of this group

    // ---- Stage 1: h1 = relu(W1 @ pose + b1), K=3 scalar; store [p][c] f16 ----
    {
        const int p   = tid & 31;       // pixel 0..31
        const int c0  = tid >> 5;       // 0..7, 8 channels per thread
        const int pix = pix0 + p;
        const float q0 = poseMap[pix];
        const float q1 = poseMap[NPIX + pix];
        const float q2 = poseMap[2 * NPIX + pix];
        #pragma unroll
        for (int r = 0; r < 8; ++r) {
            const int c = c0 + 8 * r;
            float h = W1[c * 3 + 0] * q0 + W1[c * 3 + 1] * q1 +
                      W1[c * 3 + 2] * q2 + b1[c];
            hB1[p * LD1 + c] = (_Float16)fmaxf(h, 0.0f);
        }
    }
    __syncthreads();

    // ---- Stage 2: h2 = relu(W2 @ h1 + b2): 16 M-tiles x 2 N-tiles, K=64 ----
    #pragma unroll
    for (int ti = 0; ti < 2; ++ti) {
        const int t = wv * 2 + ti;      // M tile 0..15
        CFragU acc0, acc1;
        #pragma unroll
        for (int v = 0; v < 8; ++v) { acc0.f[v] = 0.0f; acc1.f[v] = 0.0f; }
        #pragma unroll
        for (int ks = 0; ks < 2; ++ks) {
            v16h a  = load_a_frag(W2h, CCH, t * 16, ks * 32, lane);  // shared A
            v16h b0 = load_b_frag(hB1, LD1,  0, ks * 32, lane);
            v16h b1f= load_b_frag(hB1, LD1, 16, ks * 32, lane);
            acc0.v = __builtin_amdgcn_wmma_f32_16x16x32_f16(
                false, a, false, b0, (short)0, acc0.v, false, false);
            acc1.v = __builtin_amdgcn_wmma_f32_16x16x32_f16(
                false, a, false, b1f, (short)0, acc1.v, false, false);
        }
        // D layout: N = lane&15 (pixel within tile), M = v + 8*(lane>>4)
        const int p     = lane & 15;
        const int mbase = t * 16 + ((lane >> 4) << 3);
        #pragma unroll
        for (int v = 0; v < 8; ++v) {
            const float bb = b2[mbase + v];
            hB2[p        * LD2 + mbase + v] =
                (_Float16)fmaxf(acc0.f[v] + bb, 0.0f);
            hB2[(p + 16) * LD2 + mbase + v] =
                (_Float16)fmaxf(acc1.f[v] + bb, 0.0f);
        }
    }
    __syncthreads();

    // ---- Stage 3: w = W3 @ h2 + b3: 36 M-tiles x 2 N-tiles, K=256 ----------
    for (int t = wv; t < 36; t += 8) {  // uniform per wave
        CFragU acc0, acc1;
        #pragma unroll
        for (int v = 0; v < 8; ++v) { acc0.f[v] = 0.0f; acc1.f[v] = 0.0f; }
        #pragma unroll
        for (int ks = 0; ks < 8; ++ks) {
            v16h a  = load_a_frag(W3h, M2, t * 16, ks * 32, lane);   // shared A
            v16h b0 = load_b_frag(hB2, LD2,  0, ks * 32, lane);
            v16h b1f= load_b_frag(hB2, LD2, 16, ks * 32, lane);
            acc0.v = __builtin_amdgcn_wmma_f32_16x16x32_f16(
                false, a, false, b0, (short)0, acc0.v, false, false);
            acc1.v = __builtin_amdgcn_wmma_f32_16x16x32_f16(
                false, a, false, b1f, (short)0, acc1.v, false, false);
        }
        const int p     = lane & 15;
        const int mbase = t * 16 + ((lane >> 4) << 3);
        #pragma unroll
        for (int v = 0; v < 8; ++v) {
            const float bb = b3[mbase + v];
            wbuf[p        * LDW + mbase + v] = acc0.f[v] + bb;
            wbuf[(p + 16) * LDW + mbase + v] = acc1.f[v] + bb;
        }
    }
    __syncthreads();

    // ---- Stage 4: softmax over 9 taps + gather + output ---------------------
    // 32 px * 64 ch = 2048 (c,p) pairs; each thread handles 8 channels.
    #pragma unroll
    for (int r = 0; r < 8; ++r) {
        const int c   = r * 8 + (tid >> 5);  // 0..63
        const int p   = tid & 31;
        const int pix = pix0 + p;

        float w9[9];
        float mx = -1e30f;
        #pragma unroll
        for (int k = 0; k < 9; ++k) {
            w9[k] = wbuf[p * LDW + c * 9 + k];
            mx = fmaxf(mx, w9[k]);
        }
        float s = 0.0f;
        #pragma unroll
        for (int k = 0; k < 9; ++k) { w9[k] = __expf(w9[k] - mx); s += w9[k]; }
        const float inv = 1.0f / s;

        const int y0 = mapY[pix];
        const int x0 = mapX[pix];
        #pragma unroll
        for (int b = 0; b < 2; ++b) {
            float accv = 0.0f;
            #pragma unroll
            for (int k = 0; k < 9; ++k) {
                const int yy = y0 + (k / 3) - 1;   // pad=1 -> shift by -1
                const int xx = x0 + (k % 3) - 1;
                const bool ok = ((unsigned)yy < (unsigned)HI) &&
                                ((unsigned)xx < (unsigned)WI);
                float g = ok ? x[(((b * CCH + c) * HI) + yy) * WI + xx] : 0.0f;
                accv += g * w9[k] * inv;
            }
            out[(b * CCH + c) * NPIX + pix] = accv;
        }
    }
}

// ---------------------------------------------------------------------------
extern "C" void kernel_launch(void* const* d_in, const int* in_sizes, int n_in,
                              void* d_out, int out_size, void* d_ws, size_t ws_size,
                              hipStream_t stream) {
    const float* x    = (const float*)d_in[0];
    const float* pose = (const float*)d_in[1];
    const float* W1   = (const float*)d_in[2];
    const float* b1   = (const float*)d_in[3];
    const float* W2   = (const float*)d_in[4];
    const float* b2   = (const float*)d_in[5];
    const float* W3   = (const float*)d_in[6];
    const float* b3   = (const float*)d_in[7];
    const int*   mY   = (const int*)d_in[8];
    const int*   mX   = (const int*)d_in[9];
    float* out = (float*)d_out;

    // Workspace: f16 weight copies. W2h: 32KB @ 0, W3h: 288KB @ 32768 (16B aligned).
    _Float16* W2h = (_Float16*)d_ws;
    _Float16* W3h = W2h + M2 * CCH;

    kpn_convert_weights<<<(M3 * M2 + 255) / 256, 256, 0, stream>>>(W2, W3, W2h, W3h);
    kpn_fused_kernel<<<NPIX / NPX, 256, SMEM_SZ, stream>>>(
        x, pose, W1, b1, b2, b3, W2h, W3h, mY, mX, out);
}
